// MambaResidualBlock_18906446037280
// MI455X (gfx1250) — matrix operand order repaired and checked by reading, not compiled, not run on hardware
//
#include <hip/hip_runtime.h>

// ---------------------------------------------------------------------------
// Mamba2 residual block for MI455X (gfx1250, wave32, WMMA bf16 16x16x32).
// All dense matmuls (in-proj, SSD chunk einsums, out-proj, MLP) run through
// v_wmma_f32_16x16x32_bf16 with f32 accumulation; scalar/decay paths stay f32.
// Per-wave tile is 32x64 (8 accumulators = 64 VGPRs) to stay well under the
// 256-VGPR window and avoid the scratch spills seen with 64x64 tiles.
// ---------------------------------------------------------------------------

#define SL     8192      // L
#define DMODEL 512       // DIM
#define DI     1024      // D_INNER
#define NH     16        // NHEADS
#define HD     64        // HEADDIM
#define DS     64        // D_STATE
#define CONVD  1152      // CONV_DIM
#define DPROJ  2192      // D_IN_PROJ
#define NPAD   2304      // D_IN_PROJ padded to 128 multiple
#define CHK    128       // CHUNK
#define NC     64        // number of chunks
#define MLPH   1024

typedef __attribute__((ext_vector_type(16))) __bf16 v16bf;
typedef __attribute__((ext_vector_type(8)))  __bf16 v8bf;
typedef __attribute__((ext_vector_type(8)))  float  v8f;

__device__ __forceinline__ v8f wmma_bf16(v16bf a, v16bf b, v8f c) {
  return __builtin_amdgcn_wmma_f32_16x16x32_bf16(false, a, false, b, (short)0, c,
                                                 false, false);
}

// Load one 16x32 bf16 A-fragment (row-major A, lda in elements).
// lane m = lane&15 ; element e -> k = (e<8 ? 8g+e : 16+8g+e-8)  (g = lane>>4)
__device__ __forceinline__ v16bf load_a_frag(const __bf16* __restrict__ A,
                                             int lda, int row, int k0,
                                             int g) {
  const __bf16* p = A + (size_t)row * lda + k0 + g * 8;
  v8bf lo = *(const v8bf*)p;
  v8bf hi = *(const v8bf*)(p + 16);
  return __builtin_shufflevector(lo, hi, 0, 1, 2, 3, 4, 5, 6, 7, 8, 9, 10, 11,
                                 12, 13, 14, 15);
}

// Load one 32x16 bf16 B-fragment (row-major B, ldb in elements).
// lane n = lane&15 ; element e -> k = (e<8 ? 8g+e : 16+8g+e-8)  (g = lane>>4)
__device__ __forceinline__ v16bf load_b_frag(const __bf16* __restrict__ B,
                                             int ldb, int k0, int col,
                                             int g, int ar) {
  const __bf16* p = B + (size_t)k0 * ldb + col + ar;
  v16bf r;
#pragma unroll
  for (int e = 0; e < 16; e++) {
    int k = (e < 8) ? (g * 8 + e) : (16 + g * 8 + e - 8);
    r[e] = p[(size_t)k * ldb];
  }
  return r;
}

// 32x64 output tile per wave: acc += A[32xK] * B[Kx64]
__device__ __forceinline__ void wmma_tile_32x64(const __bf16* __restrict__ A,
                                                int lda,
                                                const __bf16* __restrict__ B,
                                                int ldb, int K, v8f acc[2][4],
                                                int lane) {
  const int g = lane >> 4, ar = lane & 15;
  for (int k0 = 0; k0 < K; k0 += 32) {
    v16bf af[2], bf[4];
#pragma unroll
    for (int i = 0; i < 2; i++) af[i] = load_a_frag(A, lda, i * 16 + ar, k0, g);
#pragma unroll
    for (int j = 0; j < 4; j++) bf[j] = load_b_frag(B, ldb, k0, j * 16, g, ar);
#pragma unroll
    for (int i = 0; i < 2; i++)
#pragma unroll
      for (int j = 0; j < 4; j++) acc[i][j] = wmma_bf16(af[i], bf[j], acc[i][j]);
  }
}

// ---------------------------------------------------------------------------
// 1. LayerNorm -> bf16
// ---------------------------------------------------------------------------
__global__ __launch_bounds__(256) void ln_kernel(const float* __restrict__ x,
                                                 const float* __restrict__ w,
                                                 const float* __restrict__ b,
                                                 __bf16* __restrict__ xn) {
  int t = blockIdx.x, tid = threadIdx.x;
  __shared__ float s1[256], s2[256];
  float a0 = x[(size_t)t * DMODEL + tid];
  float a1 = x[(size_t)t * DMODEL + tid + 256];
  s1[tid] = a0 + a1;
  s2[tid] = a0 * a0 + a1 * a1;
  __syncthreads();
  for (int s = 128; s > 0; s >>= 1) {
    if (tid < s) { s1[tid] += s1[tid + s]; s2[tid] += s2[tid + s]; }
    __syncthreads();
  }
  float mu = s1[0] * (1.f / DMODEL);
  float var = s2[0] * (1.f / DMODEL) - mu * mu;
  float rs = rsqrtf(var + 1e-5f);
  xn[(size_t)t * DMODEL + tid] = (__bf16)((a0 - mu) * rs * w[tid] + b[tid]);
  xn[(size_t)t * DMODEL + tid + 256] =
      (__bf16)((a1 - mu) * rs * w[tid + 256] + b[tid + 256]);
}

// ---------------------------------------------------------------------------
// 2. Convert/zero-pad a f32 weight to bf16 (row-major)
// ---------------------------------------------------------------------------
__global__ __launch_bounds__(256) void cvt_pad_kernel(const float* __restrict__ src,
                                                      __bf16* __restrict__ dst,
                                                      int rows, int scols,
                                                      int dcols) {
  int idx = blockIdx.x * 256 + threadIdx.x;
  if (idx >= rows * dcols) return;
  int r = idx / dcols, c = idx - r * dcols;
  dst[idx] = (c < scols) ? (__bf16)src[(size_t)r * scols + c] : (__bf16)0.0f;
}

// ---------------------------------------------------------------------------
// 3. In-projection GEMM: zxbcdt = xn @ W_in  (M=8192, N=2304 pad, K=512)
//    epilogue splits columns into z (bf16), xBC raw (f32), dt raw (f32)
//    Block: 256 threads = 8 waves (4 m x 2 n) covering a 128x128 tile.
// ---------------------------------------------------------------------------
__global__ __launch_bounds__(256) void gemm_in_kernel(
    const __bf16* __restrict__ A, const __bf16* __restrict__ B,
    __bf16* __restrict__ zb, float* __restrict__ xbc,
    float* __restrict__ dtraw) {
  int lane = threadIdx.x & 31, wave = threadIdx.x >> 5;
  int g = lane >> 4, ar = lane & 15;
  int mb = blockIdx.y * 128 + (wave >> 1) * 32;
  int nb = blockIdx.x * 128 + (wave & 1) * 64;
  v8f acc[2][4] = {};
  wmma_tile_32x64(A + (size_t)mb * DMODEL, DMODEL, B + nb, NPAD, DMODEL, acc, lane);
#pragma unroll
  for (int i = 0; i < 2; i++)
#pragma unroll
    for (int j = 0; j < 4; j++)
#pragma unroll
      for (int v = 0; v < 8; v++) {
        int row = mb + i * 16 + v + 8 * g;
        int col = nb + j * 16 + ar;
        float val = acc[i][j][v];
        if (col < DI) zb[(size_t)row * DI + col] = (__bf16)val;
        else if (col < DI + CONVD) xbc[(size_t)row * CONVD + (col - DI)] = val;
        else if (col < DPROJ) dtraw[(size_t)row * NH + (col - DI - CONVD)] = val;
      }
}

// ---------------------------------------------------------------------------
// 4. dt = softplus(dt_raw + dt_bias)
// ---------------------------------------------------------------------------
__global__ __launch_bounds__(256) void dt_kernel(const float* __restrict__ dtraw,
                                                 const float* __restrict__ dt_bias,
                                                 float* __restrict__ dtv) {
  int idx = blockIdx.x * 256 + threadIdx.x;
  if (idx >= SL * NH) return;
  float v = dtraw[idx] + dt_bias[idx & (NH - 1)];
  dtv[idx] = (v > 20.f) ? v : log1pf(__expf(v));
}

// ---------------------------------------------------------------------------
// 5. Causal conv (D_CONV=4) + SiLU, routed to bf16 layouts:
//    xh [t][h*64+p], xhT [h*64+p][t], B [t][n], BT [n][t], C [t][n]
// ---------------------------------------------------------------------------
__global__ __launch_bounds__(256) void conv_kernel(
    const float* __restrict__ xbc, const float* __restrict__ cw,
    const float* __restrict__ cb, __bf16* __restrict__ xh,
    __bf16* __restrict__ xhT, __bf16* __restrict__ Bb,
    __bf16* __restrict__ BT, __bf16* __restrict__ Cb) {
  size_t idx = (size_t)blockIdx.x * 256 + threadIdx.x;
  if (idx >= (size_t)SL * CONVD) return;
  int t = (int)(idx / CONVD);
  int j = (int)(idx - (size_t)t * CONVD);
  float acc = cb[j];
#pragma unroll
  for (int k = 0; k < 4; k++) {
    int tt = t - 3 + k;
    if (tt >= 0) acc += cw[j * 4 + k] * xbc[(size_t)tt * CONVD + j];
  }
  float a = acc / (1.f + __expf(-acc));  // SiLU
  if (j < DI) {
    xh[(size_t)t * DI + j] = (__bf16)a;
    xhT[(size_t)j * SL + t] = (__bf16)a;
  } else if (j < DI + DS) {
    int n = j - DI;
    Bb[(size_t)t * DS + n] = (__bf16)a;
    BT[(size_t)n * SL + t] = (__bf16)a;
  } else {
    int n = j - DI - DS;
    Cb[(size_t)t * DS + n] = (__bf16)a;
  }
}

// ---------------------------------------------------------------------------
// 6. Per-chunk cumulative sum of dA = dt * A(h)
// ---------------------------------------------------------------------------
__global__ __launch_bounds__(32) void cum_kernel(const float* __restrict__ dtv,
                                                 const float* __restrict__ A_log,
                                                 float* __restrict__ cum) {
  int c = blockIdx.x, h = threadIdx.x;
  if (h >= NH) return;
  float A = -__expf(A_log[h]);
  float run = 0.f;
  for (int q = 0; q < CHK; q++) {
    size_t t = (size_t)c * CHK + q;
    run += dtv[t * NH + h] * A;
    cum[t * NH + h] = run;
  }
}

// ---------------------------------------------------------------------------
// 7. CB[c,q,k] = C_c @ B_c^T  (per chunk: M=128, N=128, K=64; 8 waves)
// ---------------------------------------------------------------------------
__global__ __launch_bounds__(256) void cb_kernel(const __bf16* __restrict__ Cb,
                                                 const __bf16* __restrict__ BT,
                                                 float* __restrict__ CB) {
  int c = blockIdx.x;
  int lane = threadIdx.x & 31, wave = threadIdx.x >> 5;
  int g = lane >> 4, ar = lane & 15;
  int mb = (wave >> 1) * 32, nb = (wave & 1) * 64;
  v8f acc[2][4] = {};
  wmma_tile_32x64(Cb + (size_t)(c * CHK + mb) * DS, DS,
                  BT + (size_t)c * CHK + nb, SL, DS, acc, lane);
  float* out = CB + (size_t)c * CHK * CHK;
#pragma unroll
  for (int i = 0; i < 2; i++)
#pragma unroll
    for (int j = 0; j < 4; j++)
#pragma unroll
      for (int v = 0; v < 8; v++)
        out[(size_t)(mb + i * 16 + v + 8 * g) * CHK + nb + j * 16 + ar] =
            acc[i][j][v];
}

// ---------------------------------------------------------------------------
// 8. states[c,h,p,n] = sum_q xhT[p,q] * (B[q,n] * w(q))  (M=64,N=64,K=128)
//    w(q) = exp(cum_last - cum[q]) * dt[q]; 2 waves per (c,h)
// ---------------------------------------------------------------------------
__global__ __launch_bounds__(64) void states_kernel(
    const __bf16* __restrict__ xhT, const __bf16* __restrict__ Bb,
    const float* __restrict__ cum, const float* __restrict__ dtv,
    float* __restrict__ st) {
  int h = blockIdx.x, c = blockIdx.y;
  int lane = threadIdx.x & 31, wave = threadIdx.x >> 5;
  int g = lane >> 4, ar = lane & 15;
  int mb = wave * 32;
  const __bf16* A = xhT + (size_t)h * HD * SL + (size_t)c * CHK;
  const __bf16* B = Bb + (size_t)c * CHK * DS;
  float cumlast = cum[((size_t)c * CHK + CHK - 1) * NH + h];
  v8f acc[2][4] = {};
  for (int k0 = 0; k0 < CHK; k0 += 32) {
    v16bf af[2];
#pragma unroll
    for (int i = 0; i < 2; i++)
      af[i] = load_a_frag(A, SL, mb + i * 16 + ar, k0, g);
    float wq[16];
#pragma unroll
    for (int e = 0; e < 16; e++) {
      size_t t = (size_t)c * CHK + k0 + ((e < 8) ? (g * 8 + e) : (16 + g * 8 + e - 8));
      wq[e] = __expf(cumlast - cum[t * NH + h]) * dtv[t * NH + h];
    }
    v16bf bf[4];
#pragma unroll
    for (int j = 0; j < 4; j++) {
      v16bf bv = load_b_frag(B, DS, k0, j * 16, g, ar);
#pragma unroll
      for (int e = 0; e < 16; e++) bv[e] = (__bf16)((float)bv[e] * wq[e]);
      bf[j] = bv;
    }
#pragma unroll
    for (int i = 0; i < 2; i++)
#pragma unroll
      for (int j = 0; j < 4; j++) acc[i][j] = wmma_bf16(af[i], bf[j], acc[i][j]);
  }
  float* out = st + ((size_t)c * NH + h) * HD * DS;
#pragma unroll
  for (int i = 0; i < 2; i++)
#pragma unroll
    for (int j = 0; j < 4; j++)
#pragma unroll
      for (int v = 0; v < 8; v++)
        out[(size_t)(mb + i * 16 + v + 8 * g) * DS + j * 16 + ar] = acc[i][j][v];
}

// ---------------------------------------------------------------------------
// 9. Sequential inter-chunk scan; writes h_prev transposed [c,h,n,p] as bf16
// ---------------------------------------------------------------------------
__global__ __launch_bounds__(256) void scan_kernel(const float* __restrict__ st,
                                                   const float* __restrict__ cum,
                                                   __bf16* __restrict__ hp) {
  int h = blockIdx.x, tid = threadIdx.x;
  float s[16];
#pragma unroll
  for (int i = 0; i < 16; i++) s[i] = 0.f;
  for (int c = 0; c < NC; c++) {
    size_t base = ((size_t)c * NH + h) * HD * DS;
#pragma unroll
    for (int i = 0; i < 16; i++) {
      int e = tid + 256 * i;          // e = p*64 + n
      int pp = e >> 6, nn = e & 63;
      hp[base + (size_t)nn * HD + pp] = (__bf16)s[i];  // transposed [n][p]
    }
    float cd = __expf(cum[((size_t)c * CHK + CHK - 1) * NH + h]);
#pragma unroll
    for (int i = 0; i < 16; i++) s[i] = cd * s[i] + st[base + tid + 256 * i];
  }
}

// ---------------------------------------------------------------------------
// 10. Y_diag: scores(q,k) = CB*exp(cum_q-cum_k)*dt_k*[q>=k]; Y = scores@xh
//     Score A-fragments are synthesized directly in WMMA register layout.
//     Epilogue adds D[h]*xh. (per (c,h): M=128, N=64, K=128; 4 waves/block)
// ---------------------------------------------------------------------------
__global__ __launch_bounds__(128) void ydiag_kernel(
    const float* __restrict__ CB, const float* __restrict__ cum,
    const float* __restrict__ dtv, const __bf16* __restrict__ xh,
    const float* __restrict__ Dp, float* __restrict__ Y) {
  int h = blockIdx.x, c = blockIdx.y;
  int lane = threadIdx.x & 31, wave = threadIdx.x >> 5;
  int g = lane >> 4, ar = lane & 15;
  int mb = wave * 32;
  int cbase = c * CHK;
  const float* CBc = CB + (size_t)c * CHK * CHK;
  const __bf16* B = xh + (size_t)cbase * DI + h * HD;
  v8f acc[2][4] = {};
  for (int k0 = 0; k0 < CHK; k0 += 32) {
    v16bf af[2];
#pragma unroll
    for (int i = 0; i < 2; i++) {
      int q = mb + i * 16 + ar;
      float cq = cum[(size_t)(cbase + q) * NH + h];
      v16bf a;
#pragma unroll
      for (int e = 0; e < 16; e++) {
        int k = k0 + ((e < 8) ? (g * 8 + e) : (16 + g * 8 + e - 8));
        float sc = 0.f;
        if (q >= k)
          sc = CBc[(size_t)q * CHK + k] *
               __expf(cq - cum[(size_t)(cbase + k) * NH + h]) *
               dtv[(size_t)(cbase + k) * NH + h];
        a[e] = (__bf16)sc;
      }
      af[i] = a;
    }
    v16bf bf[4];
#pragma unroll
    for (int j = 0; j < 4; j++) bf[j] = load_b_frag(B, DI, k0, j * 16, g, ar);
#pragma unroll
    for (int i = 0; i < 2; i++)
#pragma unroll
      for (int j = 0; j < 4; j++) acc[i][j] = wmma_bf16(af[i], bf[j], acc[i][j]);
  }
  float Dh = Dp[h];
#pragma unroll
  for (int i = 0; i < 2; i++)
#pragma unroll
    for (int j = 0; j < 4; j++)
#pragma unroll
      for (int v = 0; v < 8; v++) {
        size_t idx = (size_t)(cbase + mb + i * 16 + v + 8 * g) * DI + h * HD +
                     j * 16 + ar;
        Y[idx] = acc[i][j][v] + Dh * (float)xh[idx];
      }
}

// ---------------------------------------------------------------------------
// 11. Y_off: Y += (C_c * exp(cum_q)) @ h_prev^T   (M=128, N=64, K=64)
// ---------------------------------------------------------------------------
__global__ __launch_bounds__(128) void yoff_kernel(
    const __bf16* __restrict__ Cb, const float* __restrict__ cum,
    const __bf16* __restrict__ hp, float* __restrict__ Y) {
  int h = blockIdx.x, c = blockIdx.y;
  int lane = threadIdx.x & 31, wave = threadIdx.x >> 5;
  int g = lane >> 4, ar = lane & 15;
  int mb = wave * 32;
  int cbase = c * CHK;
  const __bf16* A = Cb + (size_t)cbase * DS;
  const __bf16* B = hp + ((size_t)c * NH + h) * DS * HD;
  v8f acc[2][4];
#pragma unroll
  for (int i = 0; i < 2; i++)
#pragma unroll
    for (int j = 0; j < 4; j++)
#pragma unroll
      for (int v = 0; v < 8; v++)
        acc[i][j][v] = Y[(size_t)(cbase + mb + i * 16 + v + 8 * g) * DI +
                         h * HD + j * 16 + ar];
  for (int k0 = 0; k0 < DS; k0 += 32) {
    v16bf af[2];
#pragma unroll
    for (int i = 0; i < 2; i++) {
      int q = mb + i * 16 + ar;
      float s = __expf(cum[(size_t)(cbase + q) * NH + h]);
      v16bf a = load_a_frag(A, DS, q, k0, g);
#pragma unroll
      for (int e = 0; e < 16; e++) a[e] = (__bf16)((float)a[e] * s);
      af[i] = a;
    }
    v16bf bf[4];
#pragma unroll
    for (int j = 0; j < 4; j++) bf[j] = load_b_frag(B, HD, k0, j * 16, g, ar);
#pragma unroll
    for (int i = 0; i < 2; i++)
#pragma unroll
      for (int j = 0; j < 4; j++) acc[i][j] = wmma_bf16(af[i], bf[j], acc[i][j]);
  }
#pragma unroll
  for (int i = 0; i < 2; i++)
#pragma unroll
    for (int j = 0; j < 4; j++)
#pragma unroll
      for (int v = 0; v < 8; v++)
        Y[(size_t)(cbase + mb + i * 16 + v + 8 * g) * DI + h * HD + j * 16 +
          ar] = acc[i][j][v];
}

// ---------------------------------------------------------------------------
// 12. y = RMSNorm(Y * silu(z)) * norm_w  -> bf16
// ---------------------------------------------------------------------------
__global__ __launch_bounds__(256) void gate_kernel(const float* __restrict__ Y,
                                                   const __bf16* __restrict__ zb,
                                                   const float* __restrict__ nw,
                                                   __bf16* __restrict__ yb) {
  int t = blockIdx.x, tid = threadIdx.x;
  __shared__ float sred[256];
  float yv[4];
  float local = 0.f;
#pragma unroll
  for (int i = 0; i < 4; i++) {
    int d = tid + 256 * i;
    float z = (float)zb[(size_t)t * DI + d];
    float v = Y[(size_t)t * DI + d] * (z / (1.f + __expf(-z)));
    yv[i] = v;
    local += v * v;
  }
  sred[tid] = local;
  __syncthreads();
  for (int s = 128; s > 0; s >>= 1) {
    if (tid < s) sred[tid] += sred[tid + s];
    __syncthreads();
  }
  float scale = rsqrtf(sred[0] * (1.f / DI) + 1e-5f);
#pragma unroll
  for (int i = 0; i < 4; i++) {
    int d = tid + 256 * i;
    yb[(size_t)t * DI + d] = (__bf16)(yv[i] * scale * nw[d]);
  }
}

// ---------------------------------------------------------------------------
// 13. x_mamba = x + y @ W_out  (M=8192, N=512, K=1024); writes f32 + bf16
// ---------------------------------------------------------------------------
__global__ __launch_bounds__(256) void gemm_out_kernel(
    const __bf16* __restrict__ A, const __bf16* __restrict__ B,
    const float* __restrict__ x, float* __restrict__ xm,
    __bf16* __restrict__ xmb) {
  int lane = threadIdx.x & 31, wave = threadIdx.x >> 5;
  int g = lane >> 4, ar = lane & 15;
  int mb = blockIdx.y * 128 + (wave >> 1) * 32;
  int nb = blockIdx.x * 128 + (wave & 1) * 64;
  v8f acc[2][4] = {};
  wmma_tile_32x64(A + (size_t)mb * DI, DI, B + nb, DMODEL, DI, acc, lane);
#pragma unroll
  for (int i = 0; i < 2; i++)
#pragma unroll
    for (int j = 0; j < 4; j++)
#pragma unroll
      for (int v = 0; v < 8; v++) {
        int row = mb + i * 16 + v + 8 * g;
        int col = nb + j * 16 + ar;
        float val = acc[i][j][v] + x[(size_t)row * DMODEL + col];
        xm[(size_t)row * DMODEL + col] = val;
        xmb[(size_t)row * DMODEL + col] = (__bf16)val;
      }
}

// ---------------------------------------------------------------------------
// 14. h1 = gelu(x_mamba @ W1 + b1)  (M=8192, N=1024, K=512) -> bf16
// ---------------------------------------------------------------------------
__global__ __launch_bounds__(256) void gemm_mlp1_kernel(
    const __bf16* __restrict__ A, const __bf16* __restrict__ B,
    const float* __restrict__ b1, __bf16* __restrict__ h1) {
  int lane = threadIdx.x & 31, wave = threadIdx.x >> 5;
  int g = lane >> 4, ar = lane & 15;
  int mb = blockIdx.y * 128 + (wave >> 1) * 32;
  int nb = blockIdx.x * 128 + (wave & 1) * 64;
  v8f acc[2][4] = {};
  wmma_tile_32x64(A + (size_t)mb * DMODEL, DMODEL, B + nb, MLPH, DMODEL, acc, lane);
#pragma unroll
  for (int i = 0; i < 2; i++)
#pragma unroll
    for (int j = 0; j < 4; j++)
#pragma unroll
      for (int v = 0; v < 8; v++) {
        int row = mb + i * 16 + v + 8 * g;
        int col = nb + j * 16 + ar;
        float val = acc[i][j][v] + b1[col];
        float ge = 0.5f * val *
                   (1.f + tanhf(0.7978845608f * (val + 0.044715f * val * val * val)));
        h1[(size_t)row * MLPH + col] = (__bf16)ge;
      }
}

// ---------------------------------------------------------------------------
// 15. x_out = x + h1 @ W2 + b2  (M=8192, N=512, K=1024); duplicate to d_out
// ---------------------------------------------------------------------------
__global__ __launch_bounds__(256) void gemm_mlp2_kernel(
    const __bf16* __restrict__ A, const __bf16* __restrict__ B,
    const float* __restrict__ b2, const float* __restrict__ x,
    float* __restrict__ out) {
  int lane = threadIdx.x & 31, wave = threadIdx.x >> 5;
  int g = lane >> 4, ar = lane & 15;
  int mb = blockIdx.y * 128 + (wave >> 1) * 32;
  int nb = blockIdx.x * 128 + (wave & 1) * 64;
  v8f acc[2][4] = {};
  wmma_tile_32x64(A + (size_t)mb * MLPH, MLPH, B + nb, DMODEL, MLPH, acc, lane);
#pragma unroll
  for (int i = 0; i < 2; i++)
#pragma unroll
    for (int j = 0; j < 4; j++)
#pragma unroll
      for (int v = 0; v < 8; v++) {
        int row = mb + i * 16 + v + 8 * g;
        int col = nb + j * 16 + ar;
        float val = acc[i][j][v] + b2[col] + x[(size_t)row * DMODEL + col];
        out[(size_t)row * DMODEL + col] = val;
        out[(size_t)SL * DMODEL + (size_t)row * DMODEL + col] = val;
      }
}

// ---------------------------------------------------------------------------
// Host launch
// ---------------------------------------------------------------------------
extern "C" void kernel_launch(void* const* d_in, const int* in_sizes, int n_in,
                              void* d_out, int out_size, void* d_ws,
                              size_t ws_size, hipStream_t stream) {
  (void)in_sizes; (void)n_in; (void)out_size; (void)ws_size;
  const float* x       = (const float*)d_in[0];
  const float* W_in    = (const float*)d_in[1];
  const float* conv_w  = (const float*)d_in[2];
  const float* conv_b  = (const float*)d_in[3];
  const float* A_log   = (const float*)d_in[4];
  const float* dt_bias = (const float*)d_in[5];
  const float* Dp      = (const float*)d_in[6];
  const float* norm_w  = (const float*)d_in[7];
  const float* W_out   = (const float*)d_in[8];
  const float* ln_w    = (const float*)d_in[9];
  const float* ln_b    = (const float*)d_in[10];
  const float* W1      = (const float*)d_in[11];
  const float* b1      = (const float*)d_in[12];
  const float* W2      = (const float*)d_in[13];
  const float* b2      = (const float*)d_in[14];
  float* out = (float*)d_out;

  char* w = (char*)d_ws;
  size_t off = 0;
  auto take = [&](size_t bytes) {
    char* p = w + off;
    off += (bytes + 255) & ~(size_t)255;
    return (void*)p;
  };
  __bf16* xn_bf  = (__bf16*)take((size_t)SL * DMODEL * 2);
  __bf16* Wst    = (__bf16*)take((size_t)DMODEL * NPAD * 2);   // reused 4x
  __bf16* z_bf   = (__bf16*)take((size_t)SL * DI * 2);
  float*  xbc    = (float*) take((size_t)SL * CONVD * 4);
  float*  dtraw  = (float*) take((size_t)SL * NH * 4);
  float*  dtv    = (float*) take((size_t)SL * NH * 4);
  float*  cum    = (float*) take((size_t)SL * NH * 4);
  __bf16* xh_bf  = (__bf16*)take((size_t)SL * DI * 2);
  __bf16* xhT_bf = (__bf16*)take((size_t)DI * SL * 2);
  __bf16* B_bf   = (__bf16*)take((size_t)SL * DS * 2);
  __bf16* BT_bf  = (__bf16*)take((size_t)DS * SL * 2);
  __bf16* C_bf   = (__bf16*)take((size_t)SL * DS * 2);
  float*  CBbuf  = (float*) take((size_t)NC * CHK * CHK * 4);
  float*  stbuf  = (float*) take((size_t)NC * NH * HD * DS * 4);
  __bf16* hprevT = (__bf16*)take((size_t)NC * NH * HD * DS * 2);
  float*  Ybuf   = (float*) take((size_t)SL * DI * 4);
  __bf16* y_bf   = (__bf16*)take((size_t)SL * DI * 2);          // reused as h1
  float*  xm     = (float*) take((size_t)SL * DMODEL * 4);
  __bf16* xm_bf  = (__bf16*)take((size_t)SL * DMODEL * 2);

  // 1. LayerNorm
  ln_kernel<<<SL, 256, 0, stream>>>(x, ln_w, ln_b, xn_bf);
  // 2-3. In-projection
  cvt_pad_kernel<<<(DMODEL * NPAD + 255) / 256, 256, 0, stream>>>(W_in, Wst,
                                                                  DMODEL, DPROJ, NPAD);
  gemm_in_kernel<<<dim3(NPAD / 128, SL / 128), 256, 0, stream>>>(xn_bf, Wst, z_bf,
                                                                 xbc, dtraw);
  // 4. dt softplus
  dt_kernel<<<(SL * NH + 255) / 256, 256, 0, stream>>>(dtraw, dt_bias, dtv);
  // 5. conv + SiLU + layout routing
  conv_kernel<<<(int)(((size_t)SL * CONVD + 255) / 256), 256, 0, stream>>>(
      xbc, conv_w, conv_b, xh_bf, xhT_bf, B_bf, BT_bf, C_bf);
  // 6. cumsum of dA
  cum_kernel<<<NC, 32, 0, stream>>>(dtv, A_log, cum);
  // 7. CB = C @ B^T per chunk
  cb_kernel<<<NC, 256, 0, stream>>>(C_bf, BT_bf, CBbuf);
  // 8. per-chunk states
  states_kernel<<<dim3(NH, NC), 64, 0, stream>>>(xhT_bf, B_bf, cum, dtv, stbuf);
  // 9. inter-chunk scan
  scan_kernel<<<NH, 256, 0, stream>>>(stbuf, cum, hprevT);
  // 10. Y_diag (+ D*xh)
  ydiag_kernel<<<dim3(NH, NC), 128, 0, stream>>>(CBbuf, cum, dtv, xh_bf, Dp, Ybuf);
  // 11. Y_off accumulate
  yoff_kernel<<<dim3(NH, NC), 128, 0, stream>>>(C_bf, cum, hprevT, Ybuf);
  // 12. gate + RMSNorm
  gate_kernel<<<SL, 256, 0, stream>>>(Ybuf, z_bf, norm_w, y_bf);
  // 13. out-projection + residual
  cvt_pad_kernel<<<(DI * DMODEL + 255) / 256, 256, 0, stream>>>(W_out, Wst, DI,
                                                                DMODEL, DMODEL);
  gemm_out_kernel<<<dim3(DMODEL / 128, SL / 128), 256, 0, stream>>>(y_bf, Wst, x,
                                                                    xm, xm_bf);
  // 14. MLP up + gelu (h1 reuses y_bf buffer)
  cvt_pad_kernel<<<(DMODEL * MLPH + 255) / 256, 256, 0, stream>>>(W1, Wst, DMODEL,
                                                                  MLPH, MLPH);
  gemm_mlp1_kernel<<<dim3(MLPH / 128, SL / 128), 256, 0, stream>>>(xm_bf, Wst, b1,
                                                                   y_bf);
  // 15. MLP down + residual from original x, duplicated output
  cvt_pad_kernel<<<(MLPH * DMODEL + 255) / 256, 256, 0, stream>>>(W2, Wst, MLPH,
                                                                  DMODEL, DMODEL);
  gemm_mlp2_kernel<<<dim3(DMODEL / 128, SL / 128), 256, 0, stream>>>(y_bf, Wst, b2,
                                                                     x, out);
}
